// SelectedUnitsHead_82755429860021
// MI455X (gfx1250) — matrix operand order repaired and checked by reading, not compiled, not run on hardware
//
#include <hip/hip_runtime.h>
#include <hip/hip_bf16.h>
#include <math.h>

typedef __attribute__((ext_vector_type(16))) _Float16 v16h;
typedef __attribute__((ext_vector_type(8)))  float    v8f;

#define B_SZ      256
#define N_ENT     512
#define NP1       513
#define ENT_DIMC  256
#define KEY_DIMC  32
#define HIDC      32
#define GATEC     128
#define MAX_ENTC  64
#define INPUT_DIMC 1024
#define FUNC_DIMC 256
#define UT_DIMC   259
#define NEGC      1.0e9f

constexpr int suh_ilog2(int n) { return (n <= 1) ? 0 : 1 + suh_ilog2(n >> 1); }

// ---------------------------------------------------------------------------
// f16-WMMA GEMM, specialized on compile-time N (power of two):
//   out[M,N] = act( (A [+A2]) @ W + bias )
// A: [M,K] f32 row-major, converted to f16 in registers (b128 loads when
//    KTAIL==false). W: [K,N] f32, staged per 32-row K-slice into LDS f16
//    **column-major** (sBt[col][k], stride 40 halves) so each lane's 16
//    B-fragment halves are contiguous -> two ds_load_b128, zero repacking.
// Each wave owns NT consecutive 16x16 output tiles (A fragment reused).
// ROWPAD: remap output row r -> (r>>9)*513 + (r&511) to write key directly
// into the [B, N+1, 32] layout, skipping the zero end rows.
// ---------------------------------------------------------------------------
template <int N, int NT, bool KTAIL, bool ROWPAD, bool RELU>
__global__ void __launch_bounds__(256)
suh_gemm(const float* __restrict__ A, const float* __restrict__ A2,
         const float* __restrict__ W, const float* __restrict__ bias,
         float* __restrict__ out, int M, int K)
{
    constexpr int LOGN = suh_ilog2(N);
    constexpr int NGRP = N / (16 * NT);      // col-tile groups per row-tile
    constexpr int CST  = 40;                 // per-column LDS stride (halves)
    __shared__ _Float16 sBt[N * CST];        // one 32-row K-slice, col-major

    const int tid  = threadIdx.x;
    const int wave = tid >> 5;
    const int lane = tid & 31;
    const int hl   = lane >> 4;              // half-wave (K-group) select
    const int l16  = lane & 15;

    const int rowTiles = (M + 15) >> 4;
    const int wtile    = blockIdx.x * 8 + wave;
    const bool active  = wtile < rowTiles * NGRP;
    const int tm = active ? (wtile / NGRP) : 0;
    const int tg = active ? (wtile % NGRP) : 0;
    const int arow  = tm * 16 + l16;
    const int bcol0 = tg * (16 * NT) + l16;

    v8f acc[NT];
    #pragma unroll
    for (int t = 0; t < NT; ++t) acc[t] = v8f{0.f,0.f,0.f,0.f,0.f,0.f,0.f,0.f};

    for (int kk = 0; kk < K; kk += 32) {
        // ---- stage W[kk:kk+32, :] into LDS col-major as f16 ----
        // each thread: one column, 4 consecutive K values -> one ds_store_b64
        __syncthreads();
        for (int it = tid; it < 8 * N; it += 256) {
            int cc  = it & (N - 1);          // column
            int krl = (it >> LOGN) * 4;      // local k (0,4,...,28)
            int kr  = kk + krl;
            float w0 = 0.f, w1 = 0.f, w2 = 0.f, w3 = 0.f;
            if (!KTAIL) {
                w0 = W[(size_t)(kr + 0) * N + cc];
                w1 = W[(size_t)(kr + 1) * N + cc];
                w2 = W[(size_t)(kr + 2) * N + cc];
                w3 = W[(size_t)(kr + 3) * N + cc];
            } else {
                if (kr + 0 < K) w0 = W[(size_t)(kr + 0) * N + cc];
                if (kr + 1 < K) w1 = W[(size_t)(kr + 1) * N + cc];
                if (kr + 2 < K) w2 = W[(size_t)(kr + 2) * N + cc];
                if (kr + 3 < K) w3 = W[(size_t)(kr + 3) * N + cc];
            }
            _Float16 hp[4] = {(_Float16)w0, (_Float16)w1, (_Float16)w2, (_Float16)w3};
            *(float2*)&sBt[cc * CST + krl] = *(float2*)hp;   // 8B aligned
        }
        __syncthreads();

        if (active) {
            const bool rok = arow < M;
            const size_t abase = (size_t)arow * K;
            const int k0 = kk + (hl ? 16 : 0);    // elems 0..7
            const int k1 = kk + (hl ? 24 : 8);    // elems 8..15

            float av[16];
            #pragma unroll
            for (int e = 0; e < 16; ++e) av[e] = 0.f;

            if (KTAIL) {
                #pragma unroll
                for (int e = 0; e < 8; ++e) {
                    if (rok && k0 + e < K) {
                        av[e] = A[abase + k0 + e];
                        if (A2) av[e] += A2[abase + k0 + e];
                    }
                    if (rok && k1 + e < K) {
                        av[e + 8] = A[abase + k1 + e];
                        if (A2) av[e + 8] += A2[abase + k1 + e];
                    }
                }
            } else if (rok) {
                if (kk + 64 < K) __builtin_prefetch(&A[abase + kk + 64], 0, 1);
                *(float4*)&av[0]  = *(const float4*)(A + abase + k0);
                *(float4*)&av[4]  = *(const float4*)(A + abase + k0 + 4);
                *(float4*)&av[8]  = *(const float4*)(A + abase + k1);
                *(float4*)&av[12] = *(const float4*)(A + abase + k1 + 4);
                if (A2) {
                    float a2v[16];
                    *(float4*)&a2v[0]  = *(const float4*)(A2 + abase + k0);
                    *(float4*)&a2v[4]  = *(const float4*)(A2 + abase + k0 + 4);
                    *(float4*)&a2v[8]  = *(const float4*)(A2 + abase + k1);
                    *(float4*)&a2v[12] = *(const float4*)(A2 + abase + k1 + 4);
                    #pragma unroll
                    for (int e = 0; e < 16; ++e) av[e] += a2v[e];
                }
            }

            v16h a;
            #pragma unroll
            for (int e = 0; e < 16; ++e) a[e] = (_Float16)av[e];

            // B fragments: lane = column, halves hl*16..hl*16+15 contiguous
            // -> two ds_load_b128 per tile, same k order (16*hl + s) as the A fragment.
            #pragma unroll
            for (int t = 0; t < NT; ++t) {
                v16h b = *(const v16h*)(sBt + (bcol0 + t * 16) * CST + hl * 16);
                acc[t] = __builtin_amdgcn_wmma_f32_16x16x32_f16(
                             false, a, false, b, (short)0, acc[t], false, false);
            }
        }
    }

    if (active) {
        #pragma unroll
        for (int t = 0; t < NT; ++t) {
            const int ccol = tg * (16 * NT) + t * 16 + l16;
            const float bv = bias ? bias[ccol] : 0.f;
            #pragma unroll
            for (int v = 0; v < 8; ++v) {
                int r = tm * 16 + v + (hl ? 8 : 0);
                if (r < M) {
                    float val = acc[t][v] + bv;
                    if (RELU) val = fmaxf(val, 0.f);
                    size_t orow = ROWPAD ? ((size_t)(r >> 9) * NP1 + (size_t)(r & 511))
                                         : (size_t)r;
                    out[orow * N + ccol] = val;
                }
            }
        }
    }
}

// ---------------------------------------------------------------------------
// State init: mask0, units=0, done=0, h=c=0, zero the "end" rows of key.
// ---------------------------------------------------------------------------
__global__ void suh_init(const float* __restrict__ au_mask,
                         float* __restrict__ mask, float* __restrict__ units,
                         float* __restrict__ done, float* __restrict__ h,
                         float* __restrict__ c, float* __restrict__ key)
{
    int idx = blockIdx.x * blockDim.x + threadIdx.x;
    if (idx < B_SZ * NP1) {
        int b = idx / NP1, n = idx % NP1;
        mask[idx]  = (n < N_ENT) ? au_mask[b * N_ENT + n] : 1.0f;
        units[idx] = 0.f;
        if (n == N_ENT) {
            float* kr = key + ((size_t)b * NP1 + N_ENT) * KEY_DIMC;
            #pragma unroll
            for (int k = 0; k < KEY_DIMC; ++k) kr[k] = 0.f;
        }
    }
    if (idx < B_SZ) done[idx] = 0.f;
    if (idx < B_SZ * HIDC) { h[idx] = 0.f; c[idx] = 0.f; }
}

// ---------------------------------------------------------------------------
// One recurrent step, fused: LN-LSTM cell + pointer scores + Gumbel-argmax
// sample + mask/units/done update. One block (128 threads) per batch row.
// ---------------------------------------------------------------------------
__device__ __forceinline__ float suh_block_sum128(float v, float* sred, int tid)
{
    sred[tid] = v;
    __syncthreads();
    for (int off = 64; off > 0; off >>= 1) {
        if (tid < off) sred[tid] += sred[tid + off];
        __syncthreads();
    }
    float r = sred[0];
    __syncthreads();
    return r;
}

__device__ __forceinline__ unsigned suh_hash3(unsigned a, unsigned b, unsigned c)
{
    unsigned x = a * 0x9E3779B1u ^ b * 0x85EBCA77u ^ c * 0xC2B2AE3Du;
    x ^= x >> 16; x *= 0x7FEB352Du;
    x ^= x >> 15; x *= 0x846CA68Bu;
    x ^= x >> 16;
    return x;
}

__global__ void __launch_bounds__(128)
suh_step(const float* __restrict__ key,
         const float* __restrict__ wx, const float* __restrict__ wh,
         const float* __restrict__ lb,
         const float* __restrict__ gx, const float* __restrict__ bx,
         const float* __restrict__ gh, const float* __restrict__ bh,
         const float* __restrict__ temp,
         float* __restrict__ xcur, float* __restrict__ h, float* __restrict__ c,
         float* __restrict__ mask, float* __restrict__ units,
         float* __restrict__ done,
         float* __restrict__ logits, float* __restrict__ valid, int step)
{
    const int b   = blockIdx.x;
    const int tid = threadIdx.x;   // 128 threads

    __shared__ float sx[HIDC], shh[HIDC], sgate[GATEC];
    __shared__ float sred[GATEC];
    __shared__ float rbest[GATEC];
    __shared__ int   rbn[GATEC];

    if (tid < HIDC) { sx[tid] = xcur[b * HIDC + tid]; shh[tid] = h[b * HIDC + tid]; }
    __syncthreads();

    // gate pre-activations (each thread owns one of the 128 gate units)
    float ax = 0.f, ah = 0.f;
    #pragma unroll
    for (int k = 0; k < HIDC; ++k) {
        ax += sx[k]  * wx[k * GATEC + tid];
        ah += shh[k] * wh[k * GATEC + tid];
    }

    // LayerNorm over 128 for x-path and h-path separately
    const float inv = 1.0f / (float)GATEC;
    float sumx  = suh_block_sum128(ax,      sred, tid);
    float sumx2 = suh_block_sum128(ax * ax, sred, tid);
    float sumh  = suh_block_sum128(ah,      sred, tid);
    float sumh2 = suh_block_sum128(ah * ah, sred, tid);
    float mx = sumx * inv, vx = fmaxf(sumx2 * inv - mx * mx, 0.f);
    float mh = sumh * inv, vh = fmaxf(sumh2 * inv - mh * mh, 0.f);
    float lnx = (ax - mx) * rsqrtf(vx + 1e-5f) * gx[tid] + bx[tid];
    float lnh = (ah - mh) * rsqrtf(vh + 1e-5f) * gh[tid] + bh[tid];
    sgate[tid] = lnx + lnh + lb[tid];
    __syncthreads();

    // LSTM cell update (gate order i,f,o,u)
    if (tid < HIDC) {
        float ig = sgate[tid];
        float fg = sgate[HIDC + tid];
        float og = sgate[2 * HIDC + tid];
        float ug = sgate[3 * HIDC + tid];
        float si = 1.f / (1.f + expf(-ig));
        float sf = 1.f / (1.f + expf(-fg));
        float so = 1.f / (1.f + expf(-og));
        float cv = sf * c[b * HIDC + tid] + si * tanhf(ug);
        float hv = so * tanhf(cv);
        c[b * HIDC + tid]    = cv;
        h[b * HIDC + tid]    = hv;
        xcur[b * HIDC + tid] = hv;   // next step's input
        shh[tid] = hv;
    }
    __syncthreads();

    // pointer scores + Gumbel-argmax sampling
    const float T = temp[0];
    float best = -3.4e38f;
    int bestn = NP1 - 1;
    for (int n = tid; n < NP1; n += GATEC) {
        const float4* kr = (const float4*)(key + ((size_t)b * NP1 + n) * KEY_DIMC);
        float d = 0.f;
        #pragma unroll
        for (int k4 = 0; k4 < KEY_DIMC / 4; ++k4) {
            float4 kv = kr[k4];
            d += shh[k4*4+0]*kv.x + shh[k4*4+1]*kv.y + shh[k4*4+2]*kv.z + shh[k4*4+3]*kv.w;
        }
        float q = d * (1.0f / (float)KEY_DIMC);
        q -= (1.0f - mask[b * NP1 + n]) * NEGC;
        logits[((size_t)step * B_SZ + b) * NP1 + n] = q;
        float u = ((float)suh_hash3(0x2545F491u + step, b, n) + 0.5f)
                  * (1.0f / 4294967296.0f);
        float g = -logf(-logf(u));
        float s = q / T + g;
        if (s > best) { best = s; bestn = n; }
    }
    rbest[tid] = best; rbn[tid] = bestn;
    __syncthreads();
    for (int off = 64; off > 0; off >>= 1) {
        if (tid < off && rbest[tid + off] > rbest[tid]) {
            rbest[tid] = rbest[tid + off];
            rbn[tid]   = rbn[tid + off];
        }
        __syncthreads();
    }

    if (tid == 0) {
        int idx = rbn[0];
        float act = 1.0f - done[b];
        valid[(size_t)step * B_SZ + b] = act;
        if (act > 0.5f) {
            if (idx != N_ENT) {
                units[b * NP1 + idx] = 1.0f;
                mask[b * NP1 + idx]  = 0.0f;
            } else {
                done[b] = 1.0f;
            }
        }
    }
}

// ---------------------------------------------------------------------------
// Final projection: emb_sel = mean_n(units * key); out = embedding + emb_sel@W + b
// One block (256 threads) per batch row.
// ---------------------------------------------------------------------------
__global__ void __launch_bounds__(256)
suh_final(const float* __restrict__ key, const float* __restrict__ units,
          const float* __restrict__ embedding, const float* __restrict__ W,
          const float* __restrict__ bias, float* __restrict__ out)
{
    const int b = blockIdx.x, tid = threadIdx.x;
    __shared__ float part[8][KEY_DIMC];
    __shared__ float es[KEY_DIMC];

    int k = tid & 31, ch = tid >> 5;
    float s = 0.f;
    for (int n = ch; n < NP1; n += 8)
        s += units[b * NP1 + n] * key[((size_t)b * NP1 + n) * KEY_DIMC + k];
    part[ch][k] = s;
    __syncthreads();
    if (tid < KEY_DIMC) {
        float t = 0.f;
        #pragma unroll
        for (int j = 0; j < 8; ++j) t += part[j][tid];
        es[tid] = t * (1.0f / (float)NP1);
    }
    __syncthreads();

    for (int o = tid; o < INPUT_DIMC; o += 256) {
        float v = embedding[(size_t)b * INPUT_DIMC + o] + bias[o];
        #pragma unroll
        for (int kq = 0; kq < KEY_DIMC; ++kq)
            v += es[kq] * W[kq * INPUT_DIMC + o];
        out[(size_t)b * INPUT_DIMC + o] = v;
    }
}

// ---------------------------------------------------------------------------
extern "C" void kernel_launch(void* const* d_in, const int* in_sizes, int n_in,
                              void* d_out, int out_size, void* d_ws, size_t ws_size,
                              hipStream_t stream)
{
    (void)in_sizes; (void)n_in; (void)out_size; (void)ws_size;
    const float* embedding = (const float*)d_in[0];
    const float* ut_mask   = (const float*)d_in[1];
    const float* au_mask   = (const float*)d_in[2];
    const float* ent       = (const float*)d_in[3];
    const float* temp      = (const float*)d_in[4];
    const float* key_w     = (const float*)d_in[5];
    const float* key_b     = (const float*)d_in[6];
    const float* func_w    = (const float*)d_in[7];
    const float* func_b    = (const float*)d_in[8];
    const float* fc1_w     = (const float*)d_in[9];
    const float* fc1_b     = (const float*)d_in[10];
    const float* fc2_w     = (const float*)d_in[11];
    const float* fc2_b     = (const float*)d_in[12];
    const float* emb_w     = (const float*)d_in[13];
    const float* emb_b     = (const float*)d_in[14];
    const float* wx        = (const float*)d_in[15];
    const float* wh        = (const float*)d_in[16];
    const float* lb        = (const float*)d_in[17];
    const float* gx        = (const float*)d_in[18];
    const float* bx        = (const float*)d_in[19];
    const float* gh        = (const float*)d_in[20];
    const float* bh        = (const float*)d_in[21];

    // workspace layout (floats)
    float* ws    = (float*)d_ws;
    float* key   = ws;                                    // 256*513*32
    float* xcur  = key  + (size_t)B_SZ * NP1 * KEY_DIMC;  // 256*32
    float* hbuf  = xcur + B_SZ * HIDC;
    float* cbuf  = hbuf + B_SZ * HIDC;
    float* maskb = cbuf + B_SZ * HIDC;                    // 256*513
    float* doneb = maskb + B_SZ * NP1;                    // 256
    float* t1    = doneb + B_SZ;                          // 256*256
    float* t2    = t1 + B_SZ * FUNC_DIMC;                 // 256*256

    // output layout (concatenated tuple, flat)
    float* outp       = (float*)d_out;
    float* out_logits = outp;
    float* out_valid  = out_logits + (size_t)MAX_ENTC * B_SZ * NP1;
    float* out_units  = out_valid  + (size_t)MAX_ENTC * B_SZ;
    float* out_final  = out_units  + (size_t)B_SZ * NP1;

    // 1) init state (mask0, units=0, done=0, h=c=0, key end-rows=0)
    suh_init<<<(B_SZ * NP1 + 255) / 256, 256, 0, stream>>>(
        au_mask, maskb, out_units, doneb, hbuf, cbuf, key);

    // 2) key = entity_embedding @ key_fc_w + b  -> [B, N+1, 32] layout
    {
        int M = B_SZ * N_ENT, K = ENT_DIMC;
        int waves = (M / 16) * 1;                    // NGRP = 32/(16*2) = 1
        suh_gemm<KEY_DIMC, 2, false, true, false>
            <<<(waves + 7) / 8, 256, 0, stream>>>(
                ent, nullptr, key_w, key_b, key, M, K);
    }
    // 3) func_embed = relu(mask_ut @ func_fc_w + b)    (K=259 has a tail)
    {
        int M = B_SZ, K = UT_DIMC;
        int waves = (M / 16) * (FUNC_DIMC / 32);     // NGRP = 256/32 = 8
        suh_gemm<FUNC_DIMC, 2, true, false, true>
            <<<(waves + 7) / 8, 256, 0, stream>>>(
                ut_mask, nullptr, func_w, func_b, t1, M, K);
    }
    // 4) x = relu(embedding @ fc1_w + b)
    {
        int M = B_SZ, K = INPUT_DIMC;
        int waves = (M / 16) * (FUNC_DIMC / 32);
        suh_gemm<FUNC_DIMC, 2, false, false, true>
            <<<(waves + 7) / 8, 256, 0, stream>>>(
                embedding, nullptr, fc1_w, fc1_b, t2, M, K);
    }
    // 5) x0 = relu((x + func_embed) @ fc2_w + b)
    {
        int M = B_SZ, K = FUNC_DIMC;
        int waves = (M / 16) * 1;
        suh_gemm<KEY_DIMC, 2, false, false, true>
            <<<(waves + 7) / 8, 256, 0, stream>>>(
                t2, t1, fc2_w, fc2_b, xcur, M, K);
    }

    // 6) 64 recurrent pointer steps (sequential; one fused kernel each)
    for (int s = 0; s < MAX_ENTC; ++s) {
        suh_step<<<B_SZ, 128, 0, stream>>>(
            key, wx, wh, lb, gx, bx, gh, bh, temp,
            xcur, hbuf, cbuf, maskb, out_units, doneb,
            out_logits, out_valid, s);
    }

    // 7) final projection
    suh_final<<<B_SZ, 256, 0, stream>>>(
        key, out_units, embedding, emb_w, emb_b, out_final);
}
